// MoE_2911987826918
// MI455X (gfx1250) — compile-verified, matrix operand order checked by reading, not run on hardware
//
#include <hip/hip_runtime.h>
#include <hip/hip_bf16.h>
#include <math.h>

// Problem constants (from reference)
#define T_    4096
#define H_    1024
#define I_    512
#define E_    32
#define K_    8
#define G_    8
#define TG_   4
#define CAP_  2048
#define RSCALE 2.5f

typedef __bf16 bf16;
typedef __attribute__((ext_vector_type(16))) __bf16 v16bf;
typedef __attribute__((ext_vector_type(8)))  __bf16 v8bf;
typedef __attribute__((ext_vector_type(4)))  __bf16 v4bf;
typedef __attribute__((ext_vector_type(8)))  float  v8f;
typedef __attribute__((ext_vector_type(4)))  float  v4f;

// ---------------------------------------------------------------------------
// WMMA fragment loader (CDNA5 16-bit A 16x32 lane layout; B mirrored with
// N in place of M).  lane 0-15: K = k0+[0..7] and k0+16+[0..7]
//                    lane 16-31: K = k0+8+[0..7] and k0+24+[0..7]
// ---------------------------------------------------------------------------
__device__ inline v16bf load_frag_bf16(const bf16* __restrict__ rowp, int k0, int half) {
    const bf16* p = rowp + k0 + half * 8;
    v8bf lo = *(const v8bf*)(p);
    v8bf hi = *(const v8bf*)(p + 16);
    v16bf a;
#pragma unroll
    for (int j = 0; j < 8; j++) { a[j] = lo[j]; a[8 + j] = hi[j]; }
    return a;
}

__device__ inline v8f wmma_bf16(v16bf a, v16bf b, v8f c) {
    return __builtin_amdgcn_wmma_f32_16x16x32_bf16(false, a, false, b, (short)0, c,
                                                   false, false);
}

__device__ inline float silu_mul(float h1, float h3) {
    // silu(h1) * h3 with fast v_rcp_f32 (skip IEEE division refinement)
    return h1 * __builtin_amdgcn_rcpf(1.f + __expf(-h1)) * h3;
}

// ---------------------------------------------------------------------------
// 0a) zero expert counters
// ---------------------------------------------------------------------------
__global__ void zero_counts_kernel(int* __restrict__ counts) {
    if (threadIdx.x < E_) counts[threadIdx.x] = 0;
}

// ---------------------------------------------------------------------------
// 0b) bulk fp32 -> bf16 conversion (vectorized by 4), grid-stride
// ---------------------------------------------------------------------------
__global__ void cvt_bf16_kernel(const float* __restrict__ src, bf16* __restrict__ dst,
                                int n4) {
    int i = blockIdx.x * blockDim.x + threadIdx.x;
    const int stride = gridDim.x * blockDim.x;
    for (; i < n4; i += stride) {
        v4f f = *(const v4f*)(src + 4 * (size_t)i);
        v4bf o;
#pragma unroll
        for (int j = 0; j < 4; j++) o[j] = (bf16)f[j];
        *(v4bf*)(dst + 4 * (size_t)i) = o;
    }
}

// ---------------------------------------------------------------------------
// 1) gating: one wave32 per token, lane == expert
// ---------------------------------------------------------------------------
__global__ void moe_gate_kernel(const float* __restrict__ x,
                                const float* __restrict__ gate_w,
                                int* __restrict__ counts,
                                int* __restrict__ rows_token,
                                float* __restrict__ rows_weight) {
    const int t = blockIdx.x;
    const int lane = threadIdx.x;          // expert id, 0..31
    const float* xr = x + (size_t)t * H_;
    const float* gw = gate_w + (size_t)lane * H_;

    float s = 0.f;
    for (int h = 0; h < H_; h += 4) {
        v4f xv = *(const v4f*)(xr + h);
        v4f gv = *(const v4f*)(gw + h);
        s += xv[0] * gv[0] + xv[1] * gv[1] + xv[2] * gv[2] + xv[3] * gv[3];
    }
    s = 1.f / (1.f + __expf(-s));          // sigmoid score for this expert

    // group max over the 4 experts of each group (lanes 4g..4g+3)
    float gs = s;
    gs = fmaxf(gs, __shfl_xor(gs, 1, 32));
    gs = fmaxf(gs, __shfl_xor(gs, 2, 32));

    // rank of my group among the 8 group scores (ties -> lower index wins)
    const int g = lane >> 2;
    int rank_g = 0;
#pragma unroll
    for (int gg = 0; gg < G_; gg++) {
        float o = __shfl(gs, gg * 4, 32);
        rank_g += (o > gs) || (o == gs && gg < g);
    }
    float ms = (rank_g < TG_) ? s : -INFINITY;

    // top-K of 32 masked scores by rank counting
    int rank = 0;
#pragma unroll
    for (int l = 0; l < 32; l++) {
        float o = __shfl(ms, l, 32);
        rank += (o > ms) || (o == ms && l < lane);
    }
    const bool sel = (rank < K_) && (ms > -INFINITY);

    float wsel = sel ? s : 0.f;
    float tot = wsel;
    tot += __shfl_xor(tot, 1, 32);
    tot += __shfl_xor(tot, 2, 32);
    tot += __shfl_xor(tot, 4, 32);
    tot += __shfl_xor(tot, 8, 32);
    tot += __shfl_xor(tot, 16, 32);
    const float w = wsel / (tot + 1e-20f) * RSCALE;

    if (sel) {
        int slot = atomicAdd(&counts[lane], 1);
        if (slot < CAP_) {
            rows_token[lane * CAP_ + slot]  = t;
            rows_weight[lane * CAP_ + slot] = w;
        }
    }
}

// ---------------------------------------------------------------------------
// 2) shared expert, stage 1: actS = silu(x @ sw1^T) * (x @ sw3^T)   (T x I)
//    Each wave: 32 rows x 32 cols of both h1 and h3 -> 8 WMMAs per k-step
//    over 6 fragment loads.  4 waves/block tile N.
// ---------------------------------------------------------------------------
__global__ void shared_mlp1_kernel(const bf16* __restrict__ xbf,
                                   const bf16* __restrict__ sw1,
                                   const bf16* __restrict__ sw3,
                                   bf16* __restrict__ actS) {
    const int wave = threadIdx.x >> 5;
    const int lane = threadIdx.x & 31;
    const int half = lane >> 4;
    const int lm   = lane & 15;
    const int rowBase = blockIdx.x * 32;
    const int colBase = (blockIdx.y * 4 + wave) * 32;

    const bf16* aptr[2];
    aptr[0] = xbf + (size_t)(rowBase + lm) * H_;
    aptr[1] = xbf + (size_t)(rowBase + 16 + lm) * H_;
    const bf16* b1ptr[2];
    b1ptr[0] = sw1 + (size_t)(colBase + lm) * H_;
    b1ptr[1] = b1ptr[0] + (size_t)16 * H_;
    const bf16* b3ptr[2];
    b3ptr[0] = sw3 + (size_t)(colBase + lm) * H_;
    b3ptr[1] = b3ptr[0] + (size_t)16 * H_;

    v8f acc1[2][2] = {}; v8f acc3[2][2] = {};
    for (int k0 = 0; k0 < H_; k0 += 32) {
        __builtin_prefetch(b1ptr[0] + k0 + 32, 0, 3);
        __builtin_prefetch(b3ptr[0] + k0 + 32, 0, 3);
        v16bf a[2], b1[2], b3[2];
#pragma unroll
        for (int i = 0; i < 2; i++) {
            a[i]  = load_frag_bf16(aptr[i], k0, half);
            b1[i] = load_frag_bf16(b1ptr[i], k0, half);
            b3[i] = load_frag_bf16(b3ptr[i], k0, half);
        }
#pragma unroll
        for (int rt = 0; rt < 2; rt++)
#pragma unroll
            for (int ct = 0; ct < 2; ct++) {
                acc1[rt][ct] = wmma_bf16(a[rt], b1[ct], acc1[rt][ct]);
                acc3[rt][ct] = wmma_bf16(a[rt], b3[ct], acc3[rt][ct]);
            }
    }
#pragma unroll
    for (int rt = 0; rt < 2; rt++)
#pragma unroll
        for (int ct = 0; ct < 2; ct++)
#pragma unroll
            for (int r = 0; r < 8; r++) {
                const int M = rowBase + 16 * rt + r + 8 * half;
                const int N = colBase + 16 * ct + lm;
                actS[(size_t)M * I_ + N] = (bf16)silu_mul(acc1[rt][ct][r], acc3[rt][ct][r]);
            }
}

// ---------------------------------------------------------------------------
// 3) shared expert, stage 2: y = actS @ sw2^T  (T x H), plain store (init y)
//    Each wave: 32x32 tile -> 4 WMMAs over 4 fragment loads per k-step.
// ---------------------------------------------------------------------------
__global__ void shared_mlp2_kernel(const bf16* __restrict__ actS,
                                   const bf16* __restrict__ sw2,
                                   float* __restrict__ y) {
    const int wave = threadIdx.x >> 5;
    const int lane = threadIdx.x & 31;
    const int half = lane >> 4;
    const int lm   = lane & 15;
    const int rowBase = blockIdx.x * 32;
    const int colBase = (blockIdx.y * 4 + wave) * 32;

    const bf16* aptr[2];
    aptr[0] = actS + (size_t)(rowBase + lm) * I_;
    aptr[1] = aptr[0] + (size_t)16 * I_;
    const bf16* bptr[2];
    bptr[0] = sw2 + (size_t)(colBase + lm) * I_;
    bptr[1] = bptr[0] + (size_t)16 * I_;

    v8f acc[2][2] = {};
    for (int k0 = 0; k0 < I_; k0 += 32) {
        __builtin_prefetch(bptr[0] + k0 + 32, 0, 3);
        v16bf a[2], b[2];
#pragma unroll
        for (int i = 0; i < 2; i++) {
            a[i] = load_frag_bf16(aptr[i], k0, half);
            b[i] = load_frag_bf16(bptr[i], k0, half);
        }
#pragma unroll
        for (int rt = 0; rt < 2; rt++)
#pragma unroll
            for (int ct = 0; ct < 2; ct++)
                acc[rt][ct] = wmma_bf16(a[rt], b[ct], acc[rt][ct]);
    }
#pragma unroll
    for (int rt = 0; rt < 2; rt++)
#pragma unroll
        for (int ct = 0; ct < 2; ct++)
#pragma unroll
            for (int r = 0; r < 8; r++) {
                const int M = rowBase + 16 * rt + r + 8 * half;
                const int N = colBase + 16 * ct + lm;
                y[(size_t)M * H_ + N] = acc[rt][ct][r];
            }
}

// ---------------------------------------------------------------------------
// 4) expert FFN stage 1: actE[e] = silu(xe @ w1^T) * (xe @ w3^T)
//    Gathered rows; each wave 32x32 of both h1/h3: 8 WMMAs per k-step.
// ---------------------------------------------------------------------------
__global__ void expert_mlp1_kernel(const bf16* __restrict__ xbf,
                                   const bf16* __restrict__ w13,
                                   const int* __restrict__ counts,
                                   const int* __restrict__ rows_token,
                                   bf16* __restrict__ actE) {
    const int e = blockIdx.z;
    int cnt = counts[e]; cnt = cnt < CAP_ ? cnt : CAP_;
    const int rowBase = blockIdx.x * 32;
    if (rowBase >= cnt) return;

    const int wave = threadIdx.x >> 5;
    const int lane = threadIdx.x & 31;
    const int half = lane >> 4;
    const int lm   = lane & 15;
    const int colBase = (blockIdx.y * 4 + wave) * 32;

    const int r0 = rowBase + lm;
    const int r1 = rowBase + 16 + lm;
    const int tok0 = (r0 < cnt) ? rows_token[e * CAP_ + r0] : 0;
    const int tok1 = (r1 < cnt) ? rows_token[e * CAP_ + r1] : 0;
    const bf16* aptr[2];
    aptr[0] = xbf + (size_t)tok0 * H_;
    aptr[1] = xbf + (size_t)tok1 * H_;
    const bf16* b1ptr[2];
    b1ptr[0] = w13 + ((size_t)e * 2 * I_ + colBase + lm) * H_;
    b1ptr[1] = b1ptr[0] + (size_t)16 * H_;
    const bf16* b3ptr[2];
    b3ptr[0] = b1ptr[0] + (size_t)I_ * H_;
    b3ptr[1] = b3ptr[0] + (size_t)16 * H_;

    v8f acc1[2][2] = {}; v8f acc3[2][2] = {};
    for (int k0 = 0; k0 < H_; k0 += 32) {
        __builtin_prefetch(b1ptr[0] + k0 + 32, 0, 3);
        __builtin_prefetch(b3ptr[0] + k0 + 32, 0, 3);
        v16bf a[2], b1[2], b3[2];
#pragma unroll
        for (int i = 0; i < 2; i++) {
            a[i]  = load_frag_bf16(aptr[i], k0, half);
            b1[i] = load_frag_bf16(b1ptr[i], k0, half);
            b3[i] = load_frag_bf16(b3ptr[i], k0, half);
        }
#pragma unroll
        for (int rt = 0; rt < 2; rt++)
#pragma unroll
            for (int ct = 0; ct < 2; ct++) {
                acc1[rt][ct] = wmma_bf16(a[rt], b1[ct], acc1[rt][ct]);
                acc3[rt][ct] = wmma_bf16(a[rt], b3[ct], acc3[rt][ct]);
            }
    }
    bf16* out = actE + (size_t)e * CAP_ * I_;
#pragma unroll
    for (int rt = 0; rt < 2; rt++)
#pragma unroll
        for (int ct = 0; ct < 2; ct++)
#pragma unroll
            for (int r = 0; r < 8; r++) {
                const int M = rowBase + 16 * rt + r + 8 * half;
                const int N = colBase + 16 * ct + lm;
                out[(size_t)M * I_ + N] = (bf16)silu_mul(acc1[rt][ct][r], acc3[rt][ct][r]);
            }
}

// ---------------------------------------------------------------------------
// 5) expert FFN stage 2: y[token] += weight * (actE[e] @ w2[e]^T)
//    Each wave 32x32 tile: 4 WMMAs per k-step; guarded scaled atomics.
// ---------------------------------------------------------------------------
__global__ void expert_mlp2_kernel(const bf16* __restrict__ actE,
                                   const bf16* __restrict__ w2,
                                   const int* __restrict__ counts,
                                   const int* __restrict__ rows_token,
                                   const float* __restrict__ rows_weight,
                                   float* __restrict__ y) {
    const int e = blockIdx.z;
    int cnt = counts[e]; cnt = cnt < CAP_ ? cnt : CAP_;
    const int rowBase = blockIdx.x * 32;
    if (rowBase >= cnt) return;

    const int wave = threadIdx.x >> 5;
    const int lane = threadIdx.x & 31;
    const int half = lane >> 4;
    const int lm   = lane & 15;
    const int colBase = (blockIdx.y * 4 + wave) * 32;

    const bf16* aptr[2];
    aptr[0] = actE + ((size_t)e * CAP_ + rowBase + lm) * I_;
    aptr[1] = aptr[0] + (size_t)16 * I_;
    const bf16* bptr[2];
    bptr[0] = w2 + ((size_t)e * H_ + colBase + lm) * I_;
    bptr[1] = bptr[0] + (size_t)16 * I_;

    v8f acc[2][2] = {};
    for (int k0 = 0; k0 < I_; k0 += 32) {
        __builtin_prefetch(bptr[0] + k0 + 32, 0, 3);
        v16bf a[2], b[2];
#pragma unroll
        for (int i = 0; i < 2; i++) {
            a[i] = load_frag_bf16(aptr[i], k0, half);
            b[i] = load_frag_bf16(bptr[i], k0, half);
        }
#pragma unroll
        for (int rt = 0; rt < 2; rt++)
#pragma unroll
            for (int ct = 0; ct < 2; ct++)
                acc[rt][ct] = wmma_bf16(a[rt], b[ct], acc[rt][ct]);
    }
#pragma unroll
    for (int rt = 0; rt < 2; rt++)
#pragma unroll
        for (int r = 0; r < 8; r++) {
            const int M = rowBase + 16 * rt + r + 8 * half;
            if (M < cnt) {
                const int tok = rows_token[e * CAP_ + M];
                const float w = rows_weight[e * CAP_ + M];
#pragma unroll
                for (int ct = 0; ct < 2; ct++) {
                    const int N = colBase + 16 * ct + lm;
                    atomicAdd(y + (size_t)tok * H_ + N, acc[rt][ct][r] * w);
                }
            }
        }
}

// ---------------------------------------------------------------------------
// launcher
// ---------------------------------------------------------------------------
extern "C" void kernel_launch(void* const* d_in, const int* in_sizes, int n_in,
                              void* d_out, int out_size, void* d_ws, size_t ws_size,
                              hipStream_t stream) {
    const float* x      = (const float*)d_in[0];
    const float* gate_w = (const float*)d_in[1];
    const float* w13    = (const float*)d_in[2];
    const float* w2     = (const float*)d_in[3];
    const float* sw1    = (const float*)d_in[4];
    const float* sw2    = (const float*)d_in[5];
    const float* sw3    = (const float*)d_in[6];
    float* y = (float*)d_out;

    // workspace carve (~184 MB total: everything bf16 so the whole working
    // set fits in the 192 MB L2 after the one-time conversion pass)
    char* ws = (char*)d_ws;
    size_t off = 0;
    int*   counts      = (int*)(ws + off);   off += 256;
    int*   rows_token  = (int*)(ws + off);   off += (size_t)E_ * CAP_ * sizeof(int);
    float* rows_weight = (float*)(ws + off); off += (size_t)E_ * CAP_ * sizeof(float);
    bf16*  xbf         = (bf16*)(ws + off);  off += (size_t)T_ * H_ * sizeof(bf16);
    bf16*  w13bf       = (bf16*)(ws + off);  off += (size_t)E_ * 2 * I_ * H_ * sizeof(bf16);
    bf16*  w2bf        = (bf16*)(ws + off);  off += (size_t)E_ * H_ * I_ * sizeof(bf16);
    bf16*  sw1bf       = (bf16*)(ws + off);  off += (size_t)I_ * H_ * sizeof(bf16);
    bf16*  sw3bf       = (bf16*)(ws + off);  off += (size_t)I_ * H_ * sizeof(bf16);
    bf16*  sw2bf       = (bf16*)(ws + off);  off += (size_t)H_ * I_ * sizeof(bf16);
    bf16*  actE        = (bf16*)(ws + off);  off += (size_t)E_ * CAP_ * I_ * sizeof(bf16);
    bf16*  actS        = (bf16*)(ws + off);  off += (size_t)T_ * I_ * sizeof(bf16);
    (void)off; (void)ws_size; (void)in_sizes; (void)n_in; (void)out_size;

    zero_counts_kernel<<<1, 32, 0, stream>>>(counts);

    // one-time fp32 -> bf16 conversion (bandwidth-bound, single pass)
    cvt_bf16_kernel<<<2048, 256, 0, stream>>>(x,   xbf,   T_ * H_ / 4);
    cvt_bf16_kernel<<<2048, 256, 0, stream>>>(w13, w13bf, E_ * 2 * I_ * H_ / 4);
    cvt_bf16_kernel<<<2048, 256, 0, stream>>>(w2,  w2bf,  E_ * H_ * I_ / 4);
    cvt_bf16_kernel<<<512, 256, 0, stream>>>(sw1, sw1bf, I_ * H_ / 4);
    cvt_bf16_kernel<<<512, 256, 0, stream>>>(sw3, sw3bf, I_ * H_ / 4);
    cvt_bf16_kernel<<<512, 256, 0, stream>>>(sw2, sw2bf, H_ * I_ / 4);

    moe_gate_kernel<<<T_, 32, 0, stream>>>(x, gate_w, counts, rows_token, rows_weight);

    // shared expert writes y first (plain stores init the output buffer) ...
    shared_mlp1_kernel<<<dim3(T_ / 32, I_ / 128), 128, 0, stream>>>(xbf, sw1bf, sw3bf, actS);
    shared_mlp2_kernel<<<dim3(T_ / 32, H_ / 128), 128, 0, stream>>>(actS, sw2bf, y);

    // ... then expert paths atomically accumulate on top
    expert_mlp1_kernel<<<dim3(CAP_ / 32, I_ / 128, E_), 128, 0, stream>>>(
        xbf, w13bf, counts, rows_token, actE);
    expert_mlp2_kernel<<<dim3(CAP_ / 32, H_ / 128, E_), 128, 0, stream>>>(
        actE, w2bf, counts, rows_token, rows_weight, y);
}
